// ShamiAttention_90237262889109
// MI455X (gfx1250) — matrix operand order repaired and checked by reading, not compiled, not run on hardware
//
#include <hip/hip_runtime.h>

// ---------------- problem constants ----------------
#define B_   4
#define S_   2048
#define D_   2048
#define H_   16
#define DH_  128
#define NTOK (B_ * S_)          // 8192
#define HDH  (H_ * DH_)         // 2048
#define LOG2E 1.4426950408889634f

// ---------------- types ----------------
typedef __attribute__((ext_vector_type(16))) __bf16 bf16x16;
typedef __attribute__((ext_vector_type(8)))  float  f32x8;
typedef __attribute__((ext_vector_type(4)))  unsigned int u32x4;
typedef __attribute__((ext_vector_type(8)))  int   i32x8;
typedef __attribute__((ext_vector_type(4)))  int   i32x4;

union Frag { uint4 u[2]; bf16x16 v; };

__device__ __forceinline__ unsigned short f2bf_u(float f) {
    unsigned int u = __builtin_bit_cast(unsigned int, f);
    unsigned int r = u + 0x7FFFu + ((u >> 16) & 1u);   // round-to-nearest-even
    return (unsigned short)(r >> 16);
}

__device__ __forceinline__ f32x8 zero8() {
    f32x8 z;
#pragma unroll
    for (int i = 0; i < 8; ++i) z[i] = 0.0f;
    return z;
}

__device__ __forceinline__ f32x8 wmma_bf16(bf16x16 a, bf16x16 b, f32x8 c) {
    // D = A(16x32 bf16) * B(32x16 bf16) + C(16x16 f32)
    return __builtin_amdgcn_wmma_f32_16x16x32_bf16(false, a, false, b, (short)0, c,
                                                   false, false);
}

__device__ __forceinline__ float rowmax16(float x) {
#pragma unroll
    for (int m = 1; m < 16; m <<= 1) x = fmaxf(x, __shfl_xor(x, m, 32));
    return x;
}
__device__ __forceinline__ float rowsum16(float x) {
#pragma unroll
    for (int m = 1; m < 16; m <<= 1) x += __shfl_xor(x, m, 32);
    return x;
}

// ---------------- Tensor Data Mover: 2D tile global->LDS ----------------
// Builds a D# per CDNA5 ISA ch.8 (group0 128b, group1 256b) for a 2D tile of
// 2-byte elements with optional LDS row padding, and issues TENSOR_LOAD_TO_LDS.
#if __has_builtin(__builtin_amdgcn_tensor_load_to_lds)
#define USE_TDM 1
__device__ __forceinline__ void tdm_load_2d_bf16(unsigned lds_off, const void* gptr,
                                                 unsigned tile_d0, unsigned tile_d1,
                                                 unsigned long long stride0_elems,
                                                 unsigned pad_interval_code,
                                                 unsigned pad_amount_code) {
    unsigned long long ga = (unsigned long long)(uintptr_t)gptr;
    u32x4 g0;
    g0[0] = 1u;                                       // count=1, user-mode descriptor
    g0[1] = lds_off;                                  // lds_addr (bytes)
    g0[2] = (unsigned)ga;                             // global_addr[31:0]
    g0[3] = (unsigned)((ga >> 32) & 0x1FFFFFFull)     // global_addr[56:32]
          | (2u << 30);                               // type=2 ("image")
    i32x8 g1;
    // data_size=1 (2B), pad_enable, pad_interval, pad_amount
    g1[0] = (int)((1u << 16) | (1u << 20) |
                  (pad_interval_code << 22) | (pad_amount_code << 25));
    // [47:32] atomic_barrier_addr=0 ; [63:48] tensor_dim0[15:0] (= tile_d0, in-bounds)
    g1[1] = (int)((tile_d0 & 0xFFFFu) << 16);
    // [79:64] tensor_dim0[31:16]=0 ; [95:80] tensor_dim1[15:0] (= tile_d1)
    g1[2] = (int)((tile_d1 & 0xFFFFu) << 16);
    // [111:96] tensor_dim1[31:16]=0 ; [127:112] tile_dim0
    g1[3] = (int)((tile_d0 & 0xFFFFu) << 16);
    // [143:128] tile_dim1 ; [159:144] tile_dim2=0
    g1[4] = (int)(tile_d1 & 0xFFFFu);
    // [207:160] tensor_dim0_stride (48b, element units)
    g1[5] = (int)(unsigned)stride0_elems;
    g1[6] = (int)((unsigned)(stride0_elems >> 32) & 0xFFFFu);
    g1[7] = 0;
    i32x4 gz;
    gz[0] = 0; gz[1] = 0; gz[2] = 0; gz[3] = 0;
#if __clang_major__ >= 23
    i32x8 gz8;
#pragma unroll
    for (int i = 0; i < 8; ++i) gz8[i] = 0;
    __builtin_amdgcn_tensor_load_to_lds(g0, g1, gz, gz, gz8, 0);
#else
    __builtin_amdgcn_tensor_load_to_lds(g0, g1, gz, gz, 0);
#endif
}
#else
#define USE_TDM 0
#endif

// ---------------- prep: fp32 -> bf16 ----------------
__global__ __launch_bounds__(256) void cvt_f32_bf16(const float* __restrict__ in,
                                                    unsigned short* __restrict__ out,
                                                    int n) {
    int i = (blockIdx.x * 256 + threadIdx.x) * 4;
    if (i + 3 < n) {
        float4 v = *(const float4*)(in + i);
        unsigned long long pk = (unsigned long long)f2bf_u(v.x)
                              | ((unsigned long long)f2bf_u(v.y) << 16)
                              | ((unsigned long long)f2bf_u(v.z) << 32)
                              | ((unsigned long long)f2bf_u(v.w) << 48);
        *(unsigned long long*)(out + i) = pk;
    }
}

// ---------------- prep: W[k][n] f32 -> WT[n][k] bf16 ----------------
__global__ __launch_bounds__(256) void wtrans(const float* __restrict__ W,
                                              unsigned short* __restrict__ WT) {
    __shared__ float tile[32][33];
    const int t  = threadIdx.x;
    const int tx = t & 31;       // col within tile
    const int ty = t >> 5;       // 0..7
    const int bx = blockIdx.x;   // n tile
    const int by = blockIdx.y;   // k tile
#pragma unroll
    for (int i = 0; i < 4; ++i) {
        int kr = ty + i * 8;
        tile[kr][tx] = W[(size_t)(by * 32 + kr) * HDH + bx * 32 + tx];
    }
    __syncthreads();
#pragma unroll
    for (int i = 0; i < 4; ++i) {
        int wr = ty + i * 8;                         // local n
        WT[(size_t)(bx * 32 + wr) * D_ + by * 32 + tx] = f2bf_u(tile[tx][wr]);
    }
}

// ---------------- GEMM: C[M=8192][N=2048] = A[M][2048] * Bt[N][2048]^T ----------
// MODE 0: Q epilogue (RoPE + 1/sqrt(DH) scale, bf16 out, [b][s][h][d])
// MODE 1: K epilogue (RoPE, bf16 out, [b][s][h][d])
// MODE 2: V epilogue (bf16 out, transposed [b][h][d][s])
// MODE 3: final fp32 out [m][2048]
template <int MODE>
__global__ __launch_bounds__(256) void gemm128(const unsigned short* __restrict__ A,
                                               const unsigned short* __restrict__ Bt,
                                               void* __restrict__ dst,
                                               const float* __restrict__ fcos,
                                               const float* __restrict__ fsin) {
    constexpr int KDIM = 2048;
    __shared__ unsigned short As[128 * 40];   // 128 rows, stride 40 (16B-aligned rows)
    __shared__ unsigned short Bs[128 * 40];

    const int t    = threadIdx.x;
    const int lane = t & 31;
    const int wave = t >> 5;
    const int ln   = lane & 15;
    const int lh   = lane >> 4;
    const int wm   = wave >> 1;   // 0..3  -> 32 M-rows each
    const int wn   = wave & 1;    // 0..1  -> 64 N-cols each
    const int blockM = blockIdx.x * 128;
    const int blockN = blockIdx.y * 128;

    f32x8 acc[2][4];
#pragma unroll
    for (int mt = 0; mt < 2; ++mt)
#pragma unroll
        for (int nt = 0; nt < 4; ++nt) acc[mt][nt] = zero8();

    const int lrow = t >> 1;          // 0..127
    const int lcol = (t & 1) * 16;    // 0 or 16

    for (int k0 = 0; k0 < KDIM; k0 += 32) {
        const unsigned short* ga = A + (size_t)(blockM + lrow) * KDIM + k0 + lcol;
        *(uint4*)&As[lrow * 40 + lcol]     = *(const uint4*)ga;
        *(uint4*)&As[lrow * 40 + lcol + 8] = *(const uint4*)(ga + 8);
        const unsigned short* gb = Bt + (size_t)(blockN + lrow) * KDIM + k0 + lcol;
        *(uint4*)&Bs[lrow * 40 + lcol]     = *(const uint4*)gb;
        *(uint4*)&Bs[lrow * 40 + lcol + 8] = *(const uint4*)(gb + 8);
        if (k0 + 32 < KDIM) {            // keep next K-slab in flight (global_prefetch_b8)
            __builtin_prefetch(ga + 32, 0, 0);
            __builtin_prefetch(gb + 32, 0, 0);
        }
        __syncthreads();

        Frag af[2], bf[4];
#pragma unroll
        for (int mt = 0; mt < 2; ++mt) {
            int row = wm * 32 + mt * 16 + ln;
            af[mt].u[0] = *(const uint4*)&As[row * 40 + lh * 8];
            af[mt].u[1] = *(const uint4*)&As[row * 40 + 16 + lh * 8];
        }
#pragma unroll
        for (int nt = 0; nt < 4; ++nt) {
            int row = wn * 64 + nt * 16 + ln;
            bf[nt].u[0] = *(const uint4*)&Bs[row * 40 + lh * 8];
            bf[nt].u[1] = *(const uint4*)&Bs[row * 40 + 16 + lh * 8];
        }
#pragma unroll
        for (int mt = 0; mt < 2; ++mt)
#pragma unroll
            for (int nt = 0; nt < 4; ++nt)
                acc[mt][nt] = wmma_bf16(af[mt].v, bf[nt].v, acc[mt][nt]);
        __syncthreads();
    }

    // epilogue (C-layout: m = r + 8*lh within tile, n = ln within tile)
#pragma unroll
    for (int mt = 0; mt < 2; ++mt)
#pragma unroll
        for (int nt = 0; nt < 4; ++nt)
#pragma unroll
            for (int r = 0; r < 8; ++r) {
                float val = acc[mt][nt][r];
                int m = blockM + wm * 32 + mt * 16 + r + 8 * lh;
                int n = blockN + wn * 64 + nt * 16 + ln;
                if constexpr (MODE == 0 || MODE == 1) {
                    float partner = __shfl_xor(val, 1, 32);   // n^1 neighbor, same m
                    int srow = m & (S_ - 1);
                    int fi   = (n & (DH_ - 1)) >> 1;
                    float c  = fcos[srow * (DH_ / 2) + fi];
                    float sn = fsin[srow * (DH_ / 2) + fi];
                    float o  = (n & 1) ? (val * c + partner * sn)
                                       : (val * c - partner * sn);
                    if constexpr (MODE == 0) o *= 0.08838834764831845f; // 1/sqrt(128)
                    ((unsigned short*)dst)[(size_t)m * HDH + n] = f2bf_u(o);
                } else if constexpr (MODE == 2) {
                    int hh = n >> 7, d = n & 127;
                    int bb = m >> 11, srow = m & (S_ - 1);
                    ((unsigned short*)dst)[((size_t)((bb * H_ + hh) * DH_ + d)) * S_ + srow] =
                        f2bf_u(val);
                } else {
                    ((float*)dst)[(size_t)m * D_ + n] = val;
                }
            }
}

// ---------------- flash attention (causal, online softmax) ----------------
// grid: (B*H, S/128); block: 256 threads; wave w owns query rows [qs+16w, qs+16w+16)
__global__ __launch_bounds__(256) void flash_attn(const unsigned short* __restrict__ Qb,
                                                  const unsigned short* __restrict__ Kb,
                                                  const unsigned short* __restrict__ VTb,
                                                  unsigned short* __restrict__ Oa) {
    __shared__ unsigned short Ks[64 * 136];   // [key][d], stride 136 (128 + 4dw pad)
    __shared__ unsigned short Vs[128 * 72];   // [d][key], stride 72  (64 + 4dw pad)
    __shared__ unsigned short Ps[128 * 72];   // [qrow][key], stride 72

    const int t    = threadIdx.x;
    const int lane = t & 31;
    const int w    = t >> 5;
    const int ln   = lane & 15;
    const int lh   = lane >> 4;
    const int bh   = blockIdx.x;
    const int b    = bh >> 4;
    const int h    = bh & 15;
    const int qs   = blockIdx.y * 128;

    // preload Q fragments (A-layout, 4 k-steps of 32 over DH=128), direct from global
    Frag qf[4];
    {
        int qrow = qs + w * 16 + ln;
        const unsigned short* gq = Qb + ((size_t)(b * S_ + qrow) * H_ + h) * DH_;
#pragma unroll
        for (int kk = 0; kk < 4; ++kk) {
            qf[kk].u[0] = *(const uint4*)(gq + kk * 32 + lh * 8);
            qf[kk].u[1] = *(const uint4*)(gq + kk * 32 + 16 + lh * 8);
        }
    }

    f32x8 oacc[8];
#pragma unroll
    for (int dt = 0; dt < 8; ++dt) oacc[dt] = zero8();
    float mi[8], li[8];
#pragma unroll
    for (int r = 0; r < 8; ++r) { mi[r] = -3.0e38f; li[r] = 0.0f; }

    const int jend = qs + 128;
    for (int j = 0; j < jend; j += 64) {
#if USE_TDM
        // Tensor Data Mover: one wave issues both tile DMAs; TDM applies the LDS
        // row padding (4 dwords after each row) matching the padded strides.
        if (w == 0) {
            const unsigned short* gk = Kb + ((size_t)(b * S_ + j) * H_ + h) * DH_;
            tdm_load_2d_bf16((unsigned)(uintptr_t)(void*)Ks, gk,
                             /*tile_d0=*/DH_, /*tile_d1=*/64,
                             /*stride0=*/(unsigned long long)HDH,
                             /*pad_interval: 64 dwords=*/5, /*pad_amount: 4 dwords=*/3);
            const unsigned short* gv = VTb + ((size_t)((b * H_ + h) * DH_)) * S_ + j;
            tdm_load_2d_bf16((unsigned)(uintptr_t)(void*)Vs, gv,
                             /*tile_d0=*/64, /*tile_d1=*/DH_,
                             /*stride0=*/(unsigned long long)S_,
                             /*pad_interval: 32 dwords=*/4, /*pad_amount: 4 dwords=*/3);
            __builtin_amdgcn_s_wait_tensorcnt(0);
        }
#else
        {   // cooperative loads: K tile (64x128), V^T tile (128x64)
            int krow = t >> 2, kcol = (t & 3) * 32;
            const unsigned short* gk =
                Kb + ((size_t)(b * S_ + j + krow) * H_ + h) * DH_ + kcol;
#pragma unroll
            for (int q = 0; q < 4; ++q)
                *(uint4*)&Ks[krow * 136 + kcol + q * 8] = *(const uint4*)(gk + q * 8);

            int vrow = t >> 1, vcol = (t & 1) * 32;
            const unsigned short* gv =
                VTb + ((size_t)((b * H_ + h) * DH_ + vrow)) * S_ + j + vcol;
#pragma unroll
            for (int q = 0; q < 4; ++q)
                *(uint4*)&Vs[vrow * 72 + vcol + q * 8] = *(const uint4*)(gv + q * 8);
        }
#endif
        __syncthreads();

        // S = Q * K^T  (scale already folded into Q)
        f32x8 sacc[4];
#pragma unroll
        for (int nt = 0; nt < 4; ++nt) {
            sacc[nt] = zero8();
#pragma unroll
            for (int kk = 0; kk < 4; ++kk) {
                Frag bfr;
                int row = nt * 16 + ln;
                bfr.u[0] = *(const uint4*)&Ks[row * 136 + kk * 32 + lh * 8];
                bfr.u[1] = *(const uint4*)&Ks[row * 136 + kk * 32 + 16 + lh * 8];
                sacc[nt] = wmma_bf16(qf[kk].v, bfr.v, sacc[nt]);
            }
        }

        // causal mask (only needed on the last two kv blocks)
        if (j + 64 > qs) {
#pragma unroll
            for (int nt = 0; nt < 4; ++nt) {
                int kc = j + nt * 16 + ln;
#pragma unroll
                for (int r = 0; r < 8; ++r) {
                    int qr = qs + w * 16 + r + 8 * lh;
                    if (kc > qr) sacc[nt][r] = -3.0e38f;
                }
            }
        }

        // online softmax per row; stash P (bf16) in LDS in [qrow][key] layout
#pragma unroll
        for (int r = 0; r < 8; ++r) {
            float mx = fmaxf(fmaxf(sacc[0][r], sacc[1][r]),
                             fmaxf(sacc[2][r], sacc[3][r]));
            mx = rowmax16(mx);
            float mnew  = fmaxf(mi[r], mx);
            float alpha = exp2f((mi[r] - mnew) * LOG2E);
            float rs    = 0.0f;
            int prow    = w * 16 + r + 8 * lh;
#pragma unroll
            for (int nt = 0; nt < 4; ++nt) {
                float p = exp2f((sacc[nt][r] - mnew) * LOG2E);
                rs += p;
                Ps[prow * 72 + nt * 16 + ln] = f2bf_u(p);
            }
            rs    = rowsum16(rs);
            li[r] = li[r] * alpha + rs;
            mi[r] = mnew;
#pragma unroll
            for (int dt = 0; dt < 8; ++dt) oacc[dt][r] *= alpha;
        }
        __syncthreads();

        // O += P (16x64) * V (64x128): 2 k-steps of 32 keys
#pragma unroll
        for (int kk = 0; kk < 2; ++kk) {
            Frag pf;
            int prow = w * 16 + ln;
            pf.u[0] = *(const uint4*)&Ps[prow * 72 + kk * 32 + lh * 8];
            pf.u[1] = *(const uint4*)&Ps[prow * 72 + kk * 32 + 16 + lh * 8];
#pragma unroll
            for (int dt = 0; dt < 8; ++dt) {
                Frag vf;
                int vrow = dt * 16 + ln;
                vf.u[0] = *(const uint4*)&Vs[vrow * 72 + kk * 32 + lh * 8];
                vf.u[1] = *(const uint4*)&Vs[vrow * 72 + kk * 32 + 16 + lh * 8];
                oacc[dt] = wmma_bf16(pf.v, vf.v, oacc[dt]);
            }
        }
        __syncthreads();
    }

    // normalize and write O (bf16, [b][s][h][d])
#pragma unroll
    for (int r = 0; r < 8; ++r) {
        float inv = 1.0f / li[r];
        int srow  = qs + w * 16 + r + 8 * lh;
        unsigned short* go = Oa + ((size_t)(b * S_ + srow) * H_ + h) * DH_;
#pragma unroll
        for (int dt = 0; dt < 8; ++dt)
            go[dt * 16 + ln] = f2bf_u(oacc[dt][r] * inv);
    }
}

// ---------------- host ----------------
extern "C" void kernel_launch(void* const* d_in, const int* in_sizes, int n_in,
                              void* d_out, int out_size, void* d_ws, size_t ws_size,
                              hipStream_t stream) {
    const float* x    = (const float*)d_in[0];
    const float* fcos = (const float*)d_in[1];
    const float* fsin = (const float*)d_in[2];
    const float* Wq   = (const float*)d_in[3];
    const float* Wk   = (const float*)d_in[4];
    const float* Wv   = (const float*)d_in[5];
    const float* Wo   = (const float*)d_in[6];
    float* out = (float*)d_out;

    char* ws = (char*)d_ws;
    size_t off = 0;
    auto alloc = [&](size_t bytes) { char* p = ws + off; off += bytes; return p; };
    const size_t TOKB = (size_t)NTOK * D_ * 2;   // bf16 token tensor
    const size_t WB   = (size_t)D_ * HDH * 2;    // bf16 weight

    unsigned short* xb  = (unsigned short*)alloc(TOKB);
    unsigned short* WqT = (unsigned short*)alloc(WB);
    unsigned short* WkT = (unsigned short*)alloc(WB);
    unsigned short* WvT = (unsigned short*)alloc(WB);
    unsigned short* WoT = (unsigned short*)alloc(WB);
    unsigned short* Qb  = (unsigned short*)alloc(TOKB);
    unsigned short* Kb  = (unsigned short*)alloc(TOKB);
    unsigned short* VTb = (unsigned short*)alloc(TOKB);
    unsigned short* Oa  = (unsigned short*)alloc(TOKB);

    const int nx = NTOK * D_;
    cvt_f32_bf16<<<(nx / 4 + 255) / 256, 256, 0, stream>>>(x, xb, nx);

    dim3 tg(64, 64);
    wtrans<<<tg, 256, 0, stream>>>(Wq, WqT);
    wtrans<<<tg, 256, 0, stream>>>(Wk, WkT);
    wtrans<<<tg, 256, 0, stream>>>(Wv, WvT);
    wtrans<<<tg, 256, 0, stream>>>(Wo, WoT);

    dim3 gg(NTOK / 128, HDH / 128);   // 64 x 16
    gemm128<0><<<gg, 256, 0, stream>>>(xb, WqT, Qb,  fcos, fsin);
    gemm128<1><<<gg, 256, 0, stream>>>(xb, WkT, Kb,  fcos, fsin);
    gemm128<2><<<gg, 256, 0, stream>>>(xb, WvT, VTb, fcos, fsin);

    flash_attn<<<dim3(B_ * H_, S_ / 128), 256, 0, stream>>>(Qb, Kb, VTb, Oa);

    gemm128<3><<<gg, 256, 0, stream>>>(Oa, WoT, out, fcos, fsin);
}
